// Cell_2_1_35699768164368
// MI455X (gfx1250) — matrix operand order repaired and checked by reading, not compiled, not run on hardware
//
#include <hip/hip_runtime.h>
#include <math.h>

typedef _Float16 v16h __attribute__((ext_vector_type(16)));
typedef _Float16 v8h  __attribute__((ext_vector_type(8)));
typedef _Float16 v4h  __attribute__((ext_vector_type(4)));
typedef float    v8f  __attribute__((ext_vector_type(8)));
typedef float    v4f  __attribute__((ext_vector_type(4)));

#define BB 32
#define CC 256
#define SS 1024
#define SC (CC*SS)            // 262144 elements per (b, array)
#define SEC ((size_t)2*BB*SC) // elements per ws section (2 dirs/srcs)
#define NW (CC*CC)            // 65536 elements per attention weight matrix

static __device__ inline v8f wmma16(v16h a, v16h b, v8f c) {
  return __builtin_amdgcn_wmma_f32_16x16x32_f16(false, a, false, b, (short)0, c, false, false);
}

static __device__ inline v16h pack16(v8h lo, v8h hi) {
  v16h r;
#pragma unroll
  for (int i = 0; i < 8; ++i) { r[i] = lo[i]; r[i + 8] = hi[i]; }
  return r;
}

// A-fragment from f16 row-major (k contiguous): chunks at k = kb+2*hi8 and kb+2*hi8+8
static __device__ inline v16h a_frag_h16(const _Float16* row, int kb, int hi8) {
  v8h a0 = *(const v8h*)(row + kb + 2 * hi8);
  v8h a1 = *(const v8h*)(row + kb + 2 * hi8 + 8);
  return pack16(a0, a1);
}

// B-fragment: 16 contiguous halves at p (k(j) = hi*16 + j already folded into p)
static __device__ inline v16h b_frag_h16(const _Float16* p) {
  v8h lo = *(const v8h*)p, hi = *(const v8h*)(p + 8);
  return pack16(lo, hi);
}

// ---------------------------------------------------------------- router
__global__ __launch_bounds__(256) void router_kernel(
    const float* __restrict__ lidar, const float* __restrict__ hsi,
    const float* __restrict__ w1, const float* __restrict__ b1,
    const float* __restrict__ w2, const float* __restrict__ b2,
    float* __restrict__ out) {
  __shared__ float g[2 * CC];
  __shared__ float hbuf[128];
  const int b = blockIdx.x, tid = threadIdx.x;
  for (int ch = tid; ch < 2 * CC; ch += 256) {
    const float* src = (ch < CC) ? lidar : hsi;
    const int c = ch & (CC - 1);
    const float* p = src + ((size_t)b * CC + c) * SS;
    float s = 0.f;
    for (int i = 0; i < SS; ++i) s += p[i];
    g[ch] = s * (1.0f / SS);
  }
  __syncthreads();
  if (tid < 128) {
    float h = b1[tid];
    const float* wr = w1 + (size_t)tid * (2 * CC);
    for (int i = 0; i < 2 * CC; ++i) h += g[i] * wr[i];
    hbuf[tid] = fmaxf(h, 0.f);
  }
  __syncthreads();
  if (tid < 4) {
    float o = b2[tid];
    const float* wr = w2 + (size_t)tid * 128;
    for (int i = 0; i < 128; ++i) o += hbuf[i] * wr[i];
    out[(size_t)BB * CC * SS + b * 4 + tid] = 1.0f / (1.0f + expf(-o));
  }
}

// ---------------------------------------------------------------- weight f32->f16 prep
// Wh layout: [Wq 65536][Wk 65536][Wv 65536][Wc 131072].  grid 320, block 256
__global__ __launch_bounds__(256) void wcvt_kernel(
    const float* __restrict__ Wq, const float* __restrict__ Wk,
    const float* __restrict__ Wv, const float* __restrict__ Wc,
    _Float16* __restrict__ Wh) {
  const size_t i = ((size_t)blockIdx.x * 256 + threadIdx.x) * 4;
  const float* src;
  size_t off;
  if (i < NW)            { src = Wq; off = i; }
  else if (i < 2 * NW)   { src = Wk; off = i - NW; }
  else if (i < 3 * NW)   { src = Wv; off = i - 2 * NW; }
  else                   { src = Wc; off = i - 3 * NW; }
  v4f f = *(const v4f*)(src + off);
  v4h h;
#pragma unroll
  for (int e = 0; e < 4; ++e) h[e] = (_Float16)f[e];
  *(v4h*)(Wh + i) = h;
}

// ---------------------------------------------------------------- input transpose
// f32 [src][b][c][s] -> f16 [src][b][s][c].  grid (32, 8, 64), block 256
__global__ __launch_bounds__(256) void xpose_kernel(
    const float* __restrict__ lidar, const float* __restrict__ hsi,
    _Float16* __restrict__ Xt) {
  __shared__ _Float16 t[32][33];
  const int tid = threadIdx.x;
  const int src_b = blockIdx.z;                 // src*BB + b
  const int b = src_b & (BB - 1), src = src_b >> 5;
  const float* X = (src ? hsi : lidar) + (size_t)b * CC * SS;
  const int s0 = blockIdx.x * 32, c0 = blockIdx.y * 32;
  {
    const int cl = tid >> 3, sl4 = (tid & 7) * 4;
    v4f f = *(const v4f*)(X + (size_t)(c0 + cl) * SS + s0 + sl4);
#pragma unroll
    for (int e = 0; e < 4; ++e) t[sl4 + e][cl] = (_Float16)f[e];
  }
  __syncthreads();
  {
    const int sl = tid >> 3, cl4 = (tid & 7) * 4;
    v4h o;
#pragma unroll
    for (int e = 0; e < 4; ++e) o[e] = t[sl][cl4 + e];
    *(v4h*)(Xt + ((size_t)src_b * SS + s0 + sl) * CC + c0 + cl4) = o;
  }
}

// ---------------------------------------------------------------- projections (Q, K, QV fused)
// grid (8, 16, 64): x*8+wave = s-tile, y = o-tile, z&1 = dir, z>>1 = b
__global__ __launch_bounds__(256) void proj_kernel(
    const _Float16* __restrict__ Xt, const _Float16* __restrict__ Wh,
    const float* __restrict__ bq, const float* __restrict__ bk,
    const float* __restrict__ bv,
    _Float16* __restrict__ Qs, _Float16* __restrict__ Ks,
    _Float16* __restrict__ QVt) {
  const int tid = threadIdx.x, w = tid >> 5, lane = tid & 31;
  const int n16 = lane & 15, hi = lane >> 4, hi8 = hi * 8;
  const int dir = blockIdx.z & 1, b = blockIdx.z >> 1;
  const _Float16* Xq = Xt + ((size_t)(dir ? 1 : 0) * BB + b) * SC; // query/key source [s,c]
  const _Float16* Xv = Xt + ((size_t)(dir ? 0 : 1) * BB + b) * SC; // value source    [s,c]
  const int st = blockIdx.x * 8 + w, ot = blockIdx.y;
  const int s = st * 16 + n16;
  const int m = ot * 16 + n16;
  const _Float16* Wqr = Wh + (size_t)m * CC;
  const _Float16* Wkr = Wqr + NW;
  const _Float16* Wvr = Wqr + 2 * NW;

  v8f accQ = {}, accK = {}, accV = {};
#pragma unroll
  for (int kb8 = 0; kb8 < 8; ++kb8) {
    const int kb = kb8 * 32;
    v16h bqv = b_frag_h16(Xq + (size_t)s * CC + kb + hi * 16);
    v16h bvv = b_frag_h16(Xv + (size_t)s * CC + kb + hi * 16);
    accQ = wmma16(a_frag_h16(Wqr, kb, hi8), bqv, accQ);
    accK = wmma16(a_frag_h16(Wkr, kb, hi8), bqv, accK);
    accV = wmma16(a_frag_h16(Wvr, kb, hi8), bvv, accV);
  }
  const size_t base = ((size_t)dir * BB + b) * SC;
#pragma unroll
  for (int r = 0; r < 8; ++r) {
    const int o = ot * 16 + r + hi8;
    const float q = accQ[r] + bq[o];
    const float k = accK[r] + bk[o];
    const float v = accV[r] + bv[o];
    Qs[base + (size_t)s * CC + o]  = (_Float16)q;        // [s,c]
    Ks[base + (size_t)s * CC + o]  = (_Float16)k;        // [s,c]
    QVt[base + (size_t)o * SS + s] = (_Float16)(q * v);  // [c,s]
  }
}

// ---------------------------------------------------------------- flash attention
// grid (8, 32, 2): x*8+wave = 16-query tile, y = b, z = dir.  No block barriers.
__global__ __launch_bounds__(256, 1) void attn_kernel(
    const _Float16* __restrict__ Qs, const _Float16* __restrict__ Ks,
    const _Float16* __restrict__ QVt, _Float16* __restrict__ AO) {
  __shared__ _Float16 Pbuf[8 * 16 * 32];
  const int tid = threadIdx.x, w = tid >> 5, lane = tid & 31;
  const int n16 = lane & 15, hi = lane >> 4, hi8 = hi * 8;
  const size_t base = ((size_t)blockIdx.z * BB + blockIdx.y) * SC;
  const _Float16* Qp  = Qs + base;   // [s,c]
  const _Float16* Kp  = Ks + base;   // [s,c]
  const _Float16* QVp = QVt + base;  // [c,s]
  _Float16* AOp = AO + base;         // [s,c]
  const int q0 = (blockIdx.x * 8 + w) * 16;
  const int mq = q0 + n16;

  // resident Q A-fragments: contiguous f16 loads from [s,c]
  v16h qf[8];
#pragma unroll
  for (int kb8 = 0; kb8 < 8; ++kb8)
    qf[kb8] = a_frag_h16(Qp + (size_t)mq * CC, kb8 * 32, hi8);

  v8f acc[16];
#pragma unroll
  for (int ct = 0; ct < 16; ++ct) acc[ct] = (v8f){};
  float mrow[8], lrow[8];
#pragma unroll
  for (int r = 0; r < 8; ++r) { mrow[r] = -1e30f; lrow[r] = 0.f; }

  _Float16* pb = &Pbuf[w * 512];

#pragma unroll 1
  for (int kp = 0; kp < 32; ++kp) {
    const int kb = kp * 32;

    // prefetch next K / QV tiles (global_prefetch_b8)
    if (kp + 1 < 32) {
      __builtin_prefetch(Kp + (size_t)(kb + 32 + lane) * CC, 0, 1);
      __builtin_prefetch(QVp + (size_t)(lane * 8) * SS + kb + 32, 0, 1);
    }

    // scores: two 16x16 tiles (keys kb..kb+15, kb+16..kb+31), K read direct [s,c]
    v8f st0 = {}, st1 = {};
#pragma unroll
    for (int kb8 = 0; kb8 < 8; ++kb8) {
      const _Float16* k0 = Kp + (size_t)(kb + n16) * CC + kb8 * 32 + hi * 16;
      const _Float16* k1 = Kp + (size_t)(kb + 16 + n16) * CC + kb8 * 32 + hi * 16;
      st0 = wmma16(qf[kb8], b_frag_h16(k0), st0);
      st1 = wmma16(qf[kb8], b_frag_h16(k1), st1);
    }

    // online softmax over these 32 keys; write P to wave-private LDS
#pragma unroll
    for (int r = 0; r < 8; ++r) {
      float s0 = st0[r], s1 = st1[r];
      float tm = fmaxf(s0, s1);
      tm = fmaxf(tm, __shfl_xor(tm, 1, 32));
      tm = fmaxf(tm, __shfl_xor(tm, 2, 32));
      tm = fmaxf(tm, __shfl_xor(tm, 4, 32));
      tm = fmaxf(tm, __shfl_xor(tm, 8, 32));
      const float nm = fmaxf(mrow[r], tm);
      const float sc = __expf(mrow[r] - nm);
      const float p0 = __expf(s0 - nm);
      const float p1 = __expf(s1 - nm);
      float rs = p0 + p1;
      rs += __shfl_xor(rs, 1, 32);
      rs += __shfl_xor(rs, 2, 32);
      rs += __shfl_xor(rs, 4, 32);
      rs += __shfl_xor(rs, 8, 32);
      lrow[r] = lrow[r] * sc + rs;
      mrow[r] = nm;
#pragma unroll
      for (int ct = 0; ct < 16; ++ct) acc[ct][r] *= sc;
      const int qr = r + hi8;
      pb[qr * 32 + n16]      = (_Float16)p0;
      pb[qr * 32 + 16 + n16] = (_Float16)p1;
    }
    asm volatile("s_wait_dscnt 0" ::: "memory");

    // P D-layout -> A-fragment (m = query = n16, k = 32 keys)
    v16h pf;
    {
      const _Float16* pr = &pb[n16 * 32];
      v8h a0 = *(const v8h*)(pr + 2 * hi8);
      v8h a1 = *(const v8h*)(pr + 2 * hi8 + 8);
      pf = pack16(a0, a1);
    }

    // O += P @ QV  (QV in [c,s] -> contiguous B fragments)
#pragma unroll
    for (int ct = 0; ct < 16; ++ct) {
      const _Float16* qv = QVp + (size_t)(ct * 16 + n16) * SS + kb + hi * 16;
      acc[ct] = wmma16(pf, b_frag_h16(qv), acc[ct]);
    }
  }

  // normalize and store [s,c] f16
#pragma unroll
  for (int r = 0; r < 8; ++r) {
    const float inv = 1.0f / lrow[r];
    const size_t row = (size_t)(q0 + r + hi8) * CC;
#pragma unroll
    for (int ct = 0; ct < 16; ++ct)
      AOp[row + ct * 16 + n16] = (_Float16)(acc[ct][r] * inv);
  }
}

// ---------------------------------------------------------------- 1x1 conv + residual
// grid (8, 16, 32): x*8+wave = s-tile, y = o-tile, z = b
__global__ __launch_bounds__(256) void conv_kernel(
    const _Float16* __restrict__ AO, const _Float16* __restrict__ Wch,
    const float* __restrict__ cb, const float* __restrict__ xin,
    float* __restrict__ out) {
  const int tid = threadIdx.x, w = tid >> 5, lane = tid & 31;
  const int n16 = lane & 15, hi = lane >> 4, hi8 = hi * 8;
  const int st = blockIdx.x * 8 + w, ot = blockIdx.y, b = blockIdx.z;
  const int s = st * 16 + n16;
  const int m = ot * 16 + n16;
  const _Float16* lemb = AO + ((size_t)1 * BB + b) * SC;  // dir1 = l_emb (first in cat)
  const _Float16* hemb = AO + ((size_t)0 * BB + b) * SC;  // dir0 = h_emb
  const _Float16* Wr = Wch + (size_t)m * (2 * CC);

  // two independent accumulator chains (even/odd k-steps)
  v8f acc0 = {}, acc1 = {};
#pragma unroll
  for (int ks = 0; ks < 16; ks += 2) {
    const int kb0 = ks * 32, kb1 = kb0 + 32;
    const _Float16* s0 = (kb0 < CC) ? lemb : hemb;
    const _Float16* s1 = (kb1 < CC) ? lemb : hemb;
    acc0 = wmma16(a_frag_h16(Wr, kb0, hi8),
                  b_frag_h16(s0 + (size_t)s * CC + (kb0 & (CC - 1)) + hi * 16), acc0);
    acc1 = wmma16(a_frag_h16(Wr, kb1, hi8),
                  b_frag_h16(s1 + (size_t)s * CC + (kb1 & (CC - 1)) + hi * 16), acc1);
  }
#pragma unroll
  for (int r = 0; r < 8; ++r) {
    const int o = ot * 16 + r + hi8;
    const size_t idx = (size_t)b * SC + (size_t)o * SS + s;
    out[idx] = acc0[r] + acc1[r] + cb[o] + xin[idx];
  }
}

// ---------------------------------------------------------------- launch
extern "C" void kernel_launch(void* const* d_in, const int* in_sizes, int n_in,
                              void* d_out, int out_size, void* d_ws, size_t ws_size,
                              hipStream_t stream) {
  const float* lidar = (const float*)d_in[0];
  const float* hsi   = (const float*)d_in[1];
  const float* x     = (const float*)d_in[2];
  const float* Wq = (const float*)d_in[3];  const float* bq = (const float*)d_in[4];
  const float* Wk = (const float*)d_in[5];  const float* bk = (const float*)d_in[6];
  const float* Wv = (const float*)d_in[7];  const float* bv = (const float*)d_in[8];
  const float* conv_w = (const float*)d_in[9];
  const float* conv_b = (const float*)d_in[10];
  const float* r_w1 = (const float*)d_in[11]; const float* r_b1 = (const float*)d_in[12];
  const float* r_w2 = (const float*)d_in[13]; const float* r_b2 = (const float*)d_in[14];
  float* out = (float*)d_out;

  _Float16* ws  = (_Float16*)d_ws;
  _Float16* Qs  = ws;            // [2][B][S][C] f16
  _Float16* Ks  = ws + SEC;      // [2][B][S][C]
  _Float16* QVt = ws + 2 * SEC;  // [2][B][C][S]
  _Float16* Xt  = ws + 3 * SEC;  // [2][B][S][C]  (dead after proj)
  _Float16* AO  = Xt;            // [2][B][S][C]  (aliases Xt; written by attn)
  _Float16* Wh  = ws + 4 * SEC;  // [Wq|Wk|Wv|Wc] f16, 327680 elements

  router_kernel<<<dim3(BB), 256, 0, stream>>>(lidar, hsi, r_w1, r_b1, r_w2, r_b2, out);
  wcvt_kernel<<<dim3(320), 256, 0, stream>>>(Wq, Wk, Wv, conv_w, Wh);
  xpose_kernel<<<dim3(32, 8, 64), 256, 0, stream>>>(lidar, hsi, Xt);
  proj_kernel<<<dim3(8, 16, 64), 256, 0, stream>>>(Xt, Wh, bq, bk, bv, Qs, Ks, QVt);
  attn_kernel<<<dim3(8, BB, 2), 256, 0, stream>>>(Qs, Ks, QVt, AO);
  conv_kernel<<<dim3(8, 16, BB), 256, 0, stream>>>(AO, Wh + 3 * NW, conv_b, x, out);
}
